// MHA_40192303956451
// MI455X (gfx1250) — matrix operand order repaired and checked by reading, not compiled, not run on hardware
//
#include <hip/hip_runtime.h>
#include <hip/hip_bf16.h>

// ---------------------------------------------------------------------------
// Fused causal MHA for MI455X (gfx1250, wave32, WMMA).
// - All matmuls on v_wmma_f32_16x16x32_bf16 (f32 accumulate).
// - Flash-attention streaming softmax (no TxT score materialization).
// - K/V tiles staged to LDS with double-buffered global_load_async_to_lds_b128
//   (ASYNCcnt) when the toolchain exposes the builtins; sync LDS staging
//   fallback otherwise.
// - Dense GEMMs: 64x32 register tile -> 8 WMMA per 12 b128 loads.
// ---------------------------------------------------------------------------

#define DEVFN __device__ __forceinline__

#define AS1 __attribute__((address_space(1)))
#define AS3 __attribute__((address_space(3)))

#if __has_builtin(__builtin_amdgcn_global_load_async_to_lds_b128) && \
    __has_builtin(__builtin_amdgcn_s_wait_asynccnt)
#define ASYNC_OK 1
#else
#define ASYNC_OK 0
#endif

typedef __attribute__((ext_vector_type(16))) __bf16 v16bf;
typedef __attribute__((ext_vector_type(8)))  __bf16 v8bf;
typedef __attribute__((ext_vector_type(8)))  float  v8f;
typedef int v4i __attribute__((vector_size(16)));   // int4, matches builtin sig

namespace {
constexpr int kC   = 1184;            // n_embd
constexpr int kC3  = 3 * kC;          // 3552
constexpr int kT   = 2048;
constexpr int kB   = 4;
constexpr int kH   = 8;
constexpr int kDK  = 148;             // head dim
constexpr int kDP  = 160;             // head dim padded to 16
constexpr int kBH  = kB * kH;         // 32
constexpr int kM   = kB * kT;         // 8192 tokens
constexpr float kQScale = 0.08219949365f;  // 1/sqrt(148)
}

DEVFN __bf16 f2bf(float f) { return (__bf16)f; }

DEVFN v8f v8f_zero() {
  v8f z;
#pragma unroll
  for (int i = 0; i < 8; ++i) z[i] = 0.0f;
  return z;
}

// A-operand (16x32, M x K) fragment from row-major bf16 memory (rows K-contig).
// Lanes 0-15 row=lane, K=0..15; lanes 16-31 row=lane-16, K=16..31
// (contiguous, matching load_b_frag).
DEVFN v16bf load_a_frag(const __bf16* __restrict__ tile, int ld, int lane) {
  const int r = lane & 15;
  const int o = (lane >> 4) << 4;          // 0 or 16
  const __bf16* p = tile + r * ld + o;
  v8bf c0 = *(const v8bf*)(p);
  v8bf c1 = *(const v8bf*)(p + 8);
  v16bf a;
#pragma unroll
  for (int i = 0; i < 8; ++i) { a[i] = c0[i]; a[i + 8] = c1[i]; }
  return a;
}

// B-operand (32x16, K x N) fragment for NT GEMM: source row-major with rows =
// output columns (K-contig). Lane n holds column n: lanes 0-15 K=0..15,
// lanes 16-31 K=16..31, consecutive.
DEVFN v16bf load_b_frag(const __bf16* __restrict__ tile, int ld, int lane) {
  const int n = lane & 15;
  const int o = (lane >> 4) << 4;          // 0 or 16
  const __bf16* p = tile + n * ld + o;
  v8bf c0 = *(const v8bf*)(p);
  v8bf c1 = *(const v8bf*)(p + 8);
  v16bf b;
#pragma unroll
  for (int i = 0; i < 8; ++i) { b[i] = c0[i]; b[i + 8] = c1[i]; }
  return b;
}

DEVFN v8f wmma_bf16(v16bf a, v16bf b, v8f c) {
  return __builtin_amdgcn_wmma_f32_16x16x32_bf16(false, a, false, b,
                                                 (short)0, c, false, false);
}

// 16-byte global->LDS stage (async when available).
DEVFN void stage16(const __bf16* g, __bf16* l) {
#if ASYNC_OK
  __builtin_amdgcn_global_load_async_to_lds_b128(
      (AS1 v4i*)g, (AS3 v4i*)l, 0, 0);
#else
  *(v8bf*)l = *(const v8bf*)g;
#endif
}

// K tile: 32 keys x 160 elems, contiguous 10240 B in both global and LDS.
DEVFN void stage_k_tile(__bf16* __restrict__ lds, const __bf16* __restrict__ g,
                        int lane) {
#pragma unroll
  for (int i = 0; i < 20; ++i) {
    const int e = i * 256 + lane * 8;     // bf16 element index
    stage16(g + e, lds + e);
  }
}

// V^T tile: 160 rows (d) x 32 keys; global row stride kT, LDS row stride 32.
DEVFN void stage_v_tile(__bf16* __restrict__ lds, const __bf16* __restrict__ g,
                        int lane) {
  const int rsub = lane >> 2;             // 0..7
  const int ce   = (lane & 3) * 8;        // element offset within row
#pragma unroll
  for (int i = 0; i < 20; ++i) {
    const int r = i * 8 + rsub;
    stage16(g + (size_t)r * kT + ce, lds + r * 32 + ce);
  }
}

// ---------------------------------------------------------------------------
// Conversions / init
// ---------------------------------------------------------------------------
__global__ void cvt_f32_to_bf16(const float* __restrict__ src,
                                __bf16* __restrict__ dst, int n) {
  int i = blockIdx.x * blockDim.x + threadIdx.x;
  int stride = gridDim.x * blockDim.x;
  for (; i < n; i += stride) dst[i] = f2bf(src[i]);
}

__global__ void fill_bf16_zero(__bf16* __restrict__ dst, int n) {
  int i = blockIdx.x * blockDim.x + threadIdx.x;
  int stride = gridDim.x * blockDim.x;
  for (; i < n; i += stride) dst[i] = f2bf(0.0f);
}

// ---------------------------------------------------------------------------
// Kernel 1: qkv = x @ w_attn^T + b_attn, scattered to Q/K (scaled) and V^T.
// 64(M) x 32(N) register tile per wave: 8 WMMA per K=32 step.
// ---------------------------------------------------------------------------
DEVFN void scatter_qkv(v8f acc, int m0, int n0, int lane,
                       const float* __restrict__ bias,
                       __bf16* __restrict__ Q, __bf16* __restrict__ K,
                       __bf16* __restrict__ Vt) {
  const int n  = n0 + (lane & 15);
  const float bv = bias[n];
  const int sec = n / kC;                  // 0:q 1:k 2:v
  const int cc  = n - sec * kC;
  const int h   = cc / kDK;
  const int d   = cc - h * kDK;
#pragma unroll
  for (int j = 0; j < 8; ++j) {
    const int m  = m0 + ((lane >> 4) << 3) + j;   // token in [0, B*T)
    const int bb = m / kT;
    const int t  = m - bb * kT;
    const int bh = bb * kH + h;
    const float v = acc[j] + bv;
    if (sec == 0)      Q[((size_t)bh * kT + t) * kDP + d] = f2bf(v * kQScale);
    else if (sec == 1) K[((size_t)bh * kT + t) * kDP + d] = f2bf(v);
    else               Vt[((size_t)bh * kDP + d) * kT + t] = f2bf(v);
  }
}

__global__ void __launch_bounds__(32)
qkv_gemm_kernel(const __bf16* __restrict__ X, const __bf16* __restrict__ W,
                const float* __restrict__ bias,
                __bf16* __restrict__ Q, __bf16* __restrict__ K,
                __bf16* __restrict__ Vt) {
  const int lane = threadIdx.x;
  const int m0 = blockIdx.x * 64;
  const int n0 = blockIdx.y * 32;
  v8f acc[4][2];
#pragma unroll
  for (int i = 0; i < 4; ++i) { acc[i][0] = v8f_zero(); acc[i][1] = v8f_zero(); }
  const __bf16* Ar = X + (size_t)m0 * kC;
  const __bf16* W0 = W + (size_t)n0 * kC;
  const __bf16* W1 = W + (size_t)(n0 + 16) * kC;
  for (int k0 = 0; k0 < kC; k0 += 32) {
    if (k0 + 128 < kC) {
      __builtin_prefetch(Ar + (size_t)(lane * 2) * kC + k0 + 128, 0, 3);
      __builtin_prefetch(W0 + (size_t)lane * kC + k0 + 128, 0, 3);
    }
    v16bf b0 = load_b_frag(W0 + k0, kC, lane);
    v16bf b1 = load_b_frag(W1 + k0, kC, lane);
#pragma unroll
    for (int i = 0; i < 4; ++i) {
      v16bf a = load_a_frag(Ar + (size_t)(i * 16) * kC + k0, kC, lane);
      acc[i][0] = wmma_bf16(a, b0, acc[i][0]);
      acc[i][1] = wmma_bf16(a, b1, acc[i][1]);
    }
  }
#pragma unroll
  for (int i = 0; i < 4; ++i) {
    scatter_qkv(acc[i][0], m0 + i * 16, n0,      lane, bias, Q, K, Vt);
    scatter_qkv(acc[i][1], m0 + i * 16, n0 + 16, lane, bias, Q, K, Vt);
  }
}

// ---------------------------------------------------------------------------
// Kernel 2: flash attention. One wave per (b, h, 16-row q tile).
// Double-buffered async staging of 32-key K/V blocks into LDS; online softmax;
// P converts D-layout -> A-layout through 1 KB of LDS.
// ---------------------------------------------------------------------------
__global__ void __launch_bounds__(32)
flash_attn_kernel(const __bf16* __restrict__ Q, const __bf16* __restrict__ K,
                  const __bf16* __restrict__ Vt, __bf16* __restrict__ Y) {
  const int lane = threadIdx.x;
  const int bh = blockIdx.x;               // 0..31
  const int qt = blockIdx.y;               // 0..127
  const int b = bh / kH, h = bh - b * kH;
  const int q0 = qt * 16;
  const float NEG_INF = -__builtin_inff();

  const __bf16* Qp = Q  + (size_t)bh * kT * kDP + (size_t)q0 * kDP;
  const __bf16* Kp = K  + (size_t)bh * kT * kDP;
  const __bf16* Vp = Vt + (size_t)bh * kDP * kT;

  __shared__ alignas(16) __bf16 Klds[2][32 * kDP];   // 2 x 10240 B
  __shared__ alignas(16) __bf16 Vlds[2][kDP * 32];   // 2 x 10240 B
  __shared__ alignas(16) __bf16 Plds[16 * 32];       // 1024 B

  v16bf qa[5];
#pragma unroll
  for (int c = 0; c < 5; ++c) qa[c] = load_a_frag(Qp + c * 32, kDP, lane);

  v8f o[10];
#pragma unroll
  for (int f = 0; f < 10; ++f) o[f] = v8f_zero();
  float mi[8], li[8];
#pragma unroll
  for (int j = 0; j < 8; ++j) { mi[j] = NEG_INF; li[j] = 0.0f; }

  const int n = lane & 15;                 // key/col within subtile
  const int nblk = (qt >> 1) + 1;          // 32-key blocks

  stage_k_tile(Klds[0], Kp, lane);
  stage_v_tile(Vlds[0], Vp, lane);

  for (int blk = 0; blk < nblk; ++blk) {
    const int buf = blk & 1;
    const bool more = (blk + 1) < nblk;
    if (more) {                            // prefetch next block (other buffer)
      stage_k_tile(Klds[buf ^ 1], Kp + (size_t)(blk + 1) * 32 * kDP, lane);
      stage_v_tile(Vlds[buf ^ 1], Vp + (blk + 1) * 32, lane);
    }
#if ASYNC_OK
    if (more) __builtin_amdgcn_s_wait_asynccnt(40);  // 40 newest = next block
    else      __builtin_amdgcn_s_wait_asynccnt(0);
    asm volatile("" ::: "memory");
#else
    __syncthreads();
#endif
    const __bf16* Kb = Klds[buf];
    const __bf16* Vb = Vlds[buf];
    const int kt2 = blk * 2;
    const bool hasHi = (kt2 + 1) <= qt;    // uniform; EXEC stays full

    // ---- scores for two 16-key subtiles (d=160 in 5 K=32 chunks) ----
    v8f slo = v8f_zero();
#pragma unroll
    for (int c = 0; c < 5; ++c)
      slo = wmma_bf16(qa[c], load_b_frag(Kb + c * 32, kDP, lane), slo);
    v8f shi;
    if (hasHi) {
      shi = v8f_zero();
#pragma unroll
      for (int c = 0; c < 5; ++c)
        shi = wmma_bf16(qa[c], load_b_frag(Kb + 16 * kDP + c * 32, kDP, lane), shi);
    } else {
#pragma unroll
      for (int j = 0; j < 8; ++j) shi[j] = NEG_INF;
    }

    // ---- causal masking (diagonal tiles only) ----
#pragma unroll
    for (int j = 0; j < 8; ++j) {
      const int m = ((lane >> 4) << 3) + j;
      if (kt2 == qt && n > m) slo[j] = NEG_INF;
      if (hasHi && (kt2 + 1) == qt && n > m) shi[j] = NEG_INF;
    }

    // ---- online softmax ----
    float rm[8];
#pragma unroll
    for (int j = 0; j < 8; ++j) rm[j] = fmaxf(slo[j], shi[j]);
#pragma unroll
    for (int off = 1; off < 16; off <<= 1)
#pragma unroll
      for (int j = 0; j < 8; ++j)
        rm[j] = fmaxf(rm[j], __shfl_xor(rm[j], off, 32));

    float alpha[8], rs[8];
#pragma unroll
    for (int j = 0; j < 8; ++j) {
      const float nm = fmaxf(mi[j], rm[j]);
      alpha[j] = __expf(mi[j] - nm);
      mi[j] = nm;
      const float plo = __expf(slo[j] - nm);
      const float phi = __expf(shi[j] - nm);
      slo[j] = plo; shi[j] = phi;
      rs[j] = plo + phi;
    }
#pragma unroll
    for (int off = 1; off < 16; off <<= 1)
#pragma unroll
      for (int j = 0; j < 8; ++j)
        rs[j] += __shfl_xor(rs[j], off, 32);
#pragma unroll
    for (int j = 0; j < 8; ++j) li[j] = li[j] * alpha[j] + rs[j];
#pragma unroll
    for (int f = 0; f < 10; ++f)
#pragma unroll
      for (int j = 0; j < 8; ++j) o[f][j] *= alpha[j];

    // ---- P (D-layout) -> LDS -> A-layout ----
    __syncthreads();
#pragma unroll
    for (int j = 0; j < 8; ++j) {
      const int m = ((lane >> 4) << 3) + j;
      Plds[m * 32 + n]      = f2bf(slo[j]);
      Plds[m * 32 + 16 + n] = f2bf(shi[j]);
    }
    __syncthreads();
    v16bf pa = load_a_frag(Plds, 32, lane);

    // ---- O += P @ V^T : 10 d-tiles, K = 32 keys ----
#pragma unroll
    for (int f = 0; f < 10; ++f)
      o[f] = wmma_bf16(pa, load_b_frag(Vb + f * 16 * 32, 32, lane), o[f]);
  }

  // ---- normalize and store Y[token][h*148+d] ----
  float inv[8];
#pragma unroll
  for (int j = 0; j < 8; ++j) inv[j] = 1.0f / li[j];
#pragma unroll
  for (int f = 0; f < 10; ++f) {
    const int d = f * 16 + n;
    if (d < kDK) {
      const int col = h * kDK + d;
#pragma unroll
      for (int j = 0; j < 8; ++j) {
        const int row = b * kT + q0 + ((lane >> 4) << 3) + j;
        Y[(size_t)row * kC + col] = f2bf(o[f][j] * inv[j]);
      }
    }
  }
}

// ---------------------------------------------------------------------------
// Kernel 3: out = Y @ w_proj^T + b_proj (f32 output), 64x32 tile per wave.
// ---------------------------------------------------------------------------
DEVFN void store_out(v8f acc, int m0, int n0, int lane,
                     const float* __restrict__ bias, float* __restrict__ out) {
  const int n = n0 + (lane & 15);
  const float bv = bias[n];
#pragma unroll
  for (int j = 0; j < 8; ++j) {
    const int m = m0 + ((lane >> 4) << 3) + j;
    out[(size_t)m * kC + n] = acc[j] + bv;
  }
}

__global__ void __launch_bounds__(32)
out_proj_kernel(const __bf16* __restrict__ Y, const __bf16* __restrict__ W,
                const float* __restrict__ bias, float* __restrict__ out) {
  const int lane = threadIdx.x;
  const int m0 = blockIdx.x * 64;
  const int n0 = blockIdx.y * 32;
  v8f acc[4][2];
#pragma unroll
  for (int i = 0; i < 4; ++i) { acc[i][0] = v8f_zero(); acc[i][1] = v8f_zero(); }
  const __bf16* Yr = Y + (size_t)m0 * kC;
  const __bf16* W0 = W + (size_t)n0 * kC;
  const __bf16* W1 = W + (size_t)(n0 + 16) * kC;
  for (int k0 = 0; k0 < kC; k0 += 32) {
    if (k0 + 128 < kC) {
      __builtin_prefetch(Yr + (size_t)(lane * 2) * kC + k0 + 128, 0, 3);
      __builtin_prefetch(W0 + (size_t)lane * kC + k0 + 128, 0, 3);
    }
    v16bf b0 = load_b_frag(W0 + k0, kC, lane);
    v16bf b1 = load_b_frag(W1 + k0, kC, lane);
#pragma unroll
    for (int i = 0; i < 4; ++i) {
      v16bf a = load_a_frag(Yr + (size_t)(i * 16) * kC + k0, kC, lane);
      acc[i][0] = wmma_bf16(a, b0, acc[i][0]);
      acc[i][1] = wmma_bf16(a, b1, acc[i][1]);
    }
  }
#pragma unroll
  for (int i = 0; i < 4; ++i) {
    store_out(acc[i][0], m0 + i * 16, n0,      lane, bias, out);
    store_out(acc[i][1], m0 + i * 16, n0 + 16, lane, bias, out);
  }
}

// ---------------------------------------------------------------------------
// Host launch
// ---------------------------------------------------------------------------
extern "C" void kernel_launch(void* const* d_in, const int* in_sizes, int n_in,
                              void* d_out, int out_size, void* d_ws, size_t ws_size,
                              hipStream_t stream) {
  const float* x      = (const float*)d_in[0];
  const float* w_attn = (const float*)d_in[1];
  const float* b_attn = (const float*)d_in[2];
  const float* w_proj = (const float*)d_in[3];
  const float* b_proj = (const float*)d_in[4];
  float* out = (float*)d_out;

  size_t off = 0;
  auto wsalloc = [&](size_t bytes) -> void* {
    void* r = (char*)d_ws + off;
    off += (bytes + 255) & ~(size_t)255;
    return r;
  };
  __bf16* xb  = (__bf16*)wsalloc((size_t)kM  * kC  * 2);
  __bf16* wab = (__bf16*)wsalloc((size_t)kC3 * kC  * 2);
  __bf16* wpb = (__bf16*)wsalloc((size_t)kC  * kC  * 2);
  __bf16* Qb  = (__bf16*)wsalloc((size_t)kBH * kT  * kDP * 2);
  __bf16* Kb  = (__bf16*)wsalloc((size_t)kBH * kT  * kDP * 2);
  __bf16* Vtb = (__bf16*)wsalloc((size_t)kBH * kDP * kT  * 2);
  __bf16* Yb  = (__bf16*)wsalloc((size_t)kM  * kC  * 2);
  (void)ws_size; (void)in_sizes; (void)n_in; (void)out_size;

  const int TPB = 256;
  {
    int n = kM * kC;
    cvt_f32_to_bf16<<<(n + TPB - 1) / TPB, TPB, 0, stream>>>(x, xb, n);
    n = kC3 * kC;
    cvt_f32_to_bf16<<<(n + TPB - 1) / TPB, TPB, 0, stream>>>(w_attn, wab, n);
    n = kC * kC;
    cvt_f32_to_bf16<<<(n + TPB - 1) / TPB, TPB, 0, stream>>>(w_proj, wpb, n);
  }
  {
    int n = 3 * kBH * kT * kDP;   // zero Q/K/Vt incl. head-dim padding
    fill_bf16_zero<<<(n + TPB - 1) / TPB, TPB, 0, stream>>>(Qb, n);
  }
  qkv_gemm_kernel<<<dim3(kM / 64, kC3 / 32), 32, 0, stream>>>(
      xb, wab, b_attn, Qb, Kb, Vtb);
  flash_attn_kernel<<<dim3(kBH, kT / 16), 32, 0, stream>>>(Qb, Kb, Vtb, Yb);
  out_proj_kernel<<<dim3(kM / 64, kC / 32), 32, 0, stream>>>(
      Yb, wpb, b_proj, out);
}
